// Attention_19430432047265
// MI455X (gfx1250) — matrix operand order repaired and checked by reading, not compiled, not run on hardware
//
#include <hip/hip_runtime.h>
#include <hip/hip_bf16.h>
#include <math.h>

// ---------------- problem constants (match reference) ----------------
constexpr int B_   = 2;
constexpr int S_   = 2048;
constexpr int DIM_ = 4096;
constexpr int NH_  = 32;
constexpr int HD_  = 128;
constexpr long OUT_ELEMS   = (long)B_ * S_ * DIM_;            // 16,777,216
constexpr long CACHE_ELEMS = (long)4 * S_ * NH_ * HD_;        // 33,554,432

typedef __attribute__((ext_vector_type(16))) __bf16 bf16x16;
typedef __attribute__((ext_vector_type(8)))  __bf16 bf16x8;
typedef __attribute__((ext_vector_type(8)))  float  f32x8;
typedef __attribute__((ext_vector_type(4)))  unsigned u32x4;
typedef __attribute__((ext_vector_type(8)))  int      i32x8;
typedef __attribute__((ext_vector_type(4)))  int      i32x4;

// ---------------- CDNA5 TDM / async helpers ----------------
#if defined(__has_builtin)
#if __has_builtin(__builtin_amdgcn_tensor_load_to_lds)
#define USE_TDM 1
#endif
#endif
#ifndef USE_TDM
#define USE_TDM 0
#endif

// low 32 bits of a generic LDS pointer are the LDS byte address
__device__ __forceinline__ unsigned lds_addr_of(const void* p) {
  return (unsigned)(size_t)p;
}

#if USE_TDM
// TDM 2-D tile load: 128 rows x 32 bf16 (64B) per row, row stride 4096 elems,
// LDS pad: 16B after every 64B  ->  80B padded pitch (APITCH=40 bf16).
// D# bitfields per CDNA5 ISA.
__device__ __forceinline__ void tdm_load_tile_2d(unsigned lds_addr,
                                                 const void* tile_start) {
  unsigned long long ga = (unsigned long long)(size_t)tile_start;
  u32x4 g0 = {
      1u,                                           // count=1 (user descriptor)
      lds_addr,                                     // lds_addr [63:32]
      (unsigned)(ga & 0xFFFFFFFFu),                 // global_addr lo
      (unsigned)((ga >> 32) & 0x01FFFFFFu) | (2u << 30)   // addr hi | type=2
  };
  i32x8 g1 = {
      (int)((1u << 16) |                            // data_size = 2B
            (1u << 20) |                            // pad_enable
            (3u << 22) |                            // pad_interval: 16 DWORDs
            (3u << 25)),                            // pad_amount: 4 DWORDs
      (int)(4096u << 16),                           // tensor_dim0 = 4096 (lo16)
      (int)(4096u << 16),                           // dim0 hi=0 | tensor_dim1 lo16
      (int)(32u << 16),                             // dim1 hi=0 | tile_dim0 = 32
      128,                                          // tile_dim1=128, tile_dim2=0
      4096,                                         // tensor_dim0_stride lo32
      0, 0
  };
  i32x4 z4 = {0, 0, 0, 0};
#if __clang_major__ >= 23
  i32x8 z8 = {0, 0, 0, 0, 0, 0, 0, 0};
  __builtin_amdgcn_tensor_load_to_lds(g0, g1, z4, z4, z8, 0);
#else
  __builtin_amdgcn_tensor_load_to_lds(g0, g1, z4, z4, 0);
#endif
}
#else
// fallback: per-lane async DMA global->LDS (ASYNCcnt)
__device__ __forceinline__ void async_load_b128(unsigned lds_addr, unsigned gofs,
                                                const void* sbase) {
  asm volatile("global_load_async_to_lds_b128 %0, %1, %2"
               :: "v"(lds_addr), "v"(gofs), "s"(sbase)
               : "memory");
}
__device__ __forceinline__ void wait_async0() {
  asm volatile("s_wait_asynccnt 0x0" ::: "memory");
}
#endif

// ---------------- bf16 fragment load helpers ----------------
// A-layout fragment: elems [0..7] at p, [8..15] at p+16
__device__ __forceinline__ bf16x16 frag16_split(const __bf16* p) {
  bf16x8 a = *(const bf16x8*)p;
  bf16x8 b = *(const bf16x8*)(p + 16);
  bf16x16 r;
  #pragma unroll
  for (int e = 0; e < 8; ++e) { r[e] = a[e]; r[8 + e] = b[e]; }
  return r;
}
// B-layout fragment: elems [0..7] at p, [8..15] at p+16
__device__ __forceinline__ bf16x16 frag16_contig(const __bf16* p) {
  bf16x8 a = *(const bf16x8*)p;
  bf16x8 b = *(const bf16x8*)(p + 16);
  bf16x16 r;
  #pragma unroll
  for (int e = 0; e < 8; ++e) { r[e] = a[e]; r[8 + e] = b[e]; }
  return r;
}

// =====================================================================
// pack: fp32 -> bf16 straight copy (x)
// =====================================================================
__global__ __launch_bounds__(256)
void pack_bf16_kernel(const float* __restrict__ src, __bf16* __restrict__ dst) {
  long i = ((long)blockIdx.x * 256 + threadIdx.x) * 8;
  float4 a = *(const float4*)(src + i);
  float4 b = *(const float4*)(src + i + 4);
  bf16x8 o;
  o[0] = (__bf16)a.x; o[1] = (__bf16)a.y; o[2] = (__bf16)a.z; o[3] = (__bf16)a.w;
  o[4] = (__bf16)b.x; o[5] = (__bf16)b.y; o[6] = (__bf16)b.z; o[7] = (__bf16)b.w;
  *(bf16x8*)(dst + i) = o;
}

// =====================================================================
// transpose-pack: W[K,N] fp32 -> WT[N,K] bf16 (64x64 tiles through LDS)
// =====================================================================
__global__ __launch_bounds__(256)
void transpose_pack_bf16_kernel(const float* __restrict__ w,
                                __bf16* __restrict__ wT) {
  constexpr int TT = 64, TP = 72;          // pitch 72 bf16 = 144B (16B aligned)
  __shared__ __bf16 tile[TT * TP];         // tile[n][k]
  const int tid = threadIdx.x;
  const int kBase = blockIdx.y * TT;
  const int nBase = blockIdx.x * TT;
  #pragma unroll
  for (int i = 0; i < 4; ++i) {
    int idx = tid + i * 256;
    int k   = idx >> 4;
    int n4  = (idx & 15) * 4;
    float4 v = *(const float4*)(w + (long)(kBase + k) * DIM_ + nBase + n4);
    tile[(n4 + 0) * TP + k] = (__bf16)v.x;
    tile[(n4 + 1) * TP + k] = (__bf16)v.y;
    tile[(n4 + 2) * TP + k] = (__bf16)v.z;
    tile[(n4 + 3) * TP + k] = (__bf16)v.w;
  }
  __syncthreads();
  #pragma unroll
  for (int i = 0; i < 2; ++i) {
    int g  = tid + i * 256;
    int n  = g >> 3;
    int gk = (g & 7) * 8;
    bf16x8 v = *(const bf16x8*)&tile[n * TP + gk];
    *(bf16x8*)(wT + (long)(nBase + n) * DIM_ + kBase + gk) = v;
  }
}

// =====================================================================
// V transpose-pack: cache_v[b][s][h][d] fp32 -> vt[b][h][d][s] bf16
// grid: (S/64, HD/64, B*NH)
// =====================================================================
__global__ __launch_bounds__(256)
void v_transpose_pack_kernel(const float* __restrict__ cache_v,
                             __bf16* __restrict__ vt) {
  constexpr int TT = 64, TP = 72;
  __shared__ __bf16 tile[TT * TP];         // tile[d][s]
  const int tid = threadIdx.x;
  const int sBase = blockIdx.x * TT;
  const int dBase = blockIdx.y * TT;
  const int bh    = blockIdx.z;
  const int b = bh >> 5, h = bh & 31;
  const float* src = cache_v + (long)(b * S_) * DIM_ + h * HD_;
  #pragma unroll
  for (int i = 0; i < 4; ++i) {
    int idx = tid + i * 256;               // 64 s-rows x 16 float4
    int s  = idx >> 4;
    int d4 = (idx & 15) * 4;
    float4 v = *(const float4*)(src + (long)(sBase + s) * DIM_ + dBase + d4);
    tile[(d4 + 0) * TP + s] = (__bf16)v.x;
    tile[(d4 + 1) * TP + s] = (__bf16)v.y;
    tile[(d4 + 2) * TP + s] = (__bf16)v.z;
    tile[(d4 + 3) * TP + s] = (__bf16)v.w;
  }
  __syncthreads();
  __bf16* dst = vt + (long)(b * NH_ + h) * HD_ * S_;
  #pragma unroll
  for (int i = 0; i < 2; ++i) {
    int g  = tid + i * 256;
    int d  = g >> 3;
    int gs = (g & 7) * 8;
    bf16x8 v = *(const bf16x8*)&tile[d * TP + gs];
    *(bf16x8*)(dst + (long)(dBase + d) * S_ + sBase + gs) = v;
  }
}

// =====================================================================
// GEMM: C[4096,4096] = A * WT^T (bf16 in, fp32 out), tiles via TDM DMA.
// Block 256 threads (8 waves), tile 128x128, K-step 32.
// =====================================================================
#define TM 128
#define TN 128
#define TK 32
#define APITCH 40   // bf16/row: 32 data + 8 pad; 80B rows

__global__ __launch_bounds__(256)
void gemm_bf16wmma(const __bf16* __restrict__ A, const __bf16* __restrict__ WT,
                   float* __restrict__ C) {
  constexpr int NN = 4096, KK = 4096;
  __shared__ __bf16 As[TM * APITCH];   // [m][k]
  __shared__ __bf16 Wt[TN * APITCH];   // [n][k]

  const int tid  = threadIdx.x;
  const int lane = tid & 31;
  const int wave = tid >> 5;
  const int lo   = lane & 15;
  const int hi   = lane >> 4;
  const int mBase = blockIdx.y * TM;
  const int nBase = blockIdx.x * TN;
  const int mOff  = (wave & 1) * 64;
  const int nOff  = (wave >> 1) * 32;

  const unsigned asBase = lds_addr_of(As);
  const unsigned wtBase = lds_addr_of(Wt);

  f32x8 acc[4][2];
  #pragma unroll
  for (int mt = 0; mt < 4; ++mt)
    for (int nt = 0; nt < 2; ++nt)
      for (int r = 0; r < 8; ++r) acc[mt][nt][r] = 0.0f;

  for (int kb = 0; kb < KK; kb += TK) {
#if USE_TDM
    if (wave == 0) {   // one wave issues both tile descriptors
      tdm_load_tile_2d(asBase, (const char*)A + ((size_t)mBase * KK + kb) * 2);
      tdm_load_tile_2d(wtBase, (const char*)WT + ((size_t)nBase * KK + kb) * 2);
      __builtin_amdgcn_s_wait_tensorcnt(0);
    }
    __syncthreads();
#else
    #pragma unroll
    for (int i = 0; i < 2; ++i) {
      int g   = tid + i * 256;
      int row = g >> 2;
      int gi  = g & 3;
      unsigned l = (unsigned)(row * (APITCH * 2) + gi * 16);
      unsigned goA = (unsigned)(((mBase + row) * KK + kb) * 2 + gi * 16);
      unsigned goW = (unsigned)(((nBase + row) * KK + kb) * 2 + gi * 16);
      async_load_b128(asBase + l, goA, A);
      async_load_b128(wtBase + l, goW, WT);
      if (kb + TK < KK) {
        __builtin_prefetch(A + (mBase + row) * KK + kb + TK, 0, 1);
        __builtin_prefetch(WT + (nBase + row) * KK + kb + TK, 0, 1);
      }
    }
    wait_async0();
    __syncthreads();
#endif

    bf16x16 af[4], bfm[2];
    #pragma unroll
    for (int mt = 0; mt < 4; ++mt)
      af[mt] = frag16_split(&As[(mOff + mt * 16 + lo) * APITCH + hi * 8]);
    #pragma unroll
    for (int nt = 0; nt < 2; ++nt)
      bfm[nt] = frag16_contig(&Wt[(nOff + nt * 16 + lo) * APITCH + hi * 8]);
    #pragma unroll
    for (int mt = 0; mt < 4; ++mt)
      #pragma unroll
      for (int nt = 0; nt < 2; ++nt)
        acc[mt][nt] = __builtin_amdgcn_wmma_f32_16x16x32_bf16(
            false, af[mt], false, bfm[nt], (short)0, acc[mt][nt], false, false);
    __syncthreads();
  }

  float* cb = C + (size_t)(mBase + mOff + 8 * hi) * NN + (nBase + nOff + lo);
  #pragma unroll
  for (int mt = 0; mt < 4; ++mt)
    #pragma unroll
    for (int nt = 0; nt < 2; ++nt)
      #pragma unroll
      for (int r = 0; r < 8; ++r)
        cb[(size_t)(mt * 16 + r) * NN + nt * 16] = acc[mt][nt][r];
}

// =====================================================================
// Zero-fill cache rows for b = 2,3
// =====================================================================
__global__ __launch_bounds__(256)
void zero_tail_kernel(float* __restrict__ cache_k, float* __restrict__ cache_v) {
  long i = ((long)blockIdx.x * 256 + threadIdx.x) * 4;
  const long ofs = (long)2 * S_ * DIM_;
  float4 z = {0.f, 0.f, 0.f, 0.f};
  *(float4*)(cache_k + ofs + i) = z;
  *(float4*)(cache_v + ofs + i) = z;
}

// =====================================================================
// RoPE: q -> qb (bf16, 1/sqrt(hd) folded), k -> cache_k (fp32) + kb (bf16)
// =====================================================================
__global__ void rope_cache_kernel(const float* __restrict__ xq,
                                  const float* __restrict__ xk,
                                  const float* __restrict__ freqs,   // (S,64,2)
                                  float* __restrict__ cache_k,
                                  __bf16* __restrict__ qb,
                                  __bf16* __restrict__ kb) {
  long idx = (long)blockIdx.x * blockDim.x + threadIdx.x;
  if (idx >= (long)B_ * S_ * NH_ * (HD_ / 2)) return;
  int  d2 = idx & 63;
  int  h  = (int)((idx >> 6) & 31);
  long s  = (idx >> 11) & 2047;
  long b  = idx >> 22;
  const float scale = 0.08838834764831845f;   // 1/sqrt(128)
  float c  = freqs[(s * 64 + d2) * 2 + 0];
  float sn = freqs[(s * 64 + d2) * 2 + 1];
  long base = (b * S_ + s) * DIM_ + h * HD_ + 2 * d2;
  float qr = xq[base], qi = xq[base + 1];
  qb[base]     = (__bf16)((qr * c - qi * sn) * scale);
  qb[base + 1] = (__bf16)((qr * sn + qi * c) * scale);
  float kr = xk[base], ki = xk[base + 1];
  float kro = kr * c - ki * sn;
  float kio = kr * sn + ki * c;
  cache_k[base]     = kro;
  cache_k[base + 1] = kio;
  kb[base]     = (__bf16)kro;
  kb[base + 1] = (__bf16)kio;
}

// =====================================================================
// Flash attention: all operands pre-staged bf16; every fragment is two
// contiguous 16B loads. Output bf16 feeds the final GEMM directly.
// =====================================================================
#define PPITCH 40

__global__ __launch_bounds__(128)
void flash_attn_kernel(const __bf16* __restrict__ qb,
                       const __bf16* __restrict__ kb,
                       const __bf16* __restrict__ vt,   // [b][h][d][s]
                       __bf16* __restrict__ attnb) {
  __shared__ __bf16 Pt[4][16 * PPITCH];

  const int tid  = threadIdx.x;
  const int wave = tid >> 5;
  const int lane = tid & 31;
  const int lo   = lane & 15;
  const int hi   = lane >> 4;

  int task = blockIdx.x * 4 + wave;
  int qt = task & 127;
  int h  = (task >> 7) & 31;
  int b  = task >> 12;
  const int qBase = qt * 16;

  bf16x16 aq[4];
  const __bf16* qrow = qb + ((long)(b * S_ + qBase + lo)) * DIM_ + h * HD_;
  #pragma unroll
  for (int f = 0; f < 4; ++f)
    aq[f] = frag16_split(qrow + f * 32 + hi * 8);

  float m[8], l[8];
  f32x8 o[8];
  #pragma unroll
  for (int r = 0; r < 8; ++r) { m[r] = -__builtin_inff(); l[r] = 0.0f; }
  #pragma unroll
  for (int dt = 0; dt < 8; ++dt)
    for (int r = 0; r < 8; ++r) o[dt][r] = 0.0f;

  __bf16* P = &Pt[wave][0];
  const __bf16* vplane = vt + (long)(b * NH_ + h) * HD_ * S_;

  for (int jb = 0; jb <= qBase + 15; jb += 32) {
    f32x8 s0, s1;
    for (int r = 0; r < 8; ++r) { s0[r] = 0.0f; s1[r] = 0.0f; }
    #pragma unroll
    for (int half = 0; half < 2; ++half) {
      const __bf16* krow =
          kb + ((long)(b * S_ + jb + half * 16 + lo)) * DIM_ + h * HD_;
      #pragma unroll
      for (int f = 0; f < 4; ++f) {
        bf16x16 bk = frag16_contig(krow + f * 32 + hi * 8);
        if (half == 0)
          s0 = __builtin_amdgcn_wmma_f32_16x16x32_bf16(
              false, aq[f], false, bk, (short)0, s0, false, false);
        else
          s1 = __builtin_amdgcn_wmma_f32_16x16x32_bf16(
              false, aq[f], false, bk, (short)0, s1, false, false);
      }
    }
    #pragma unroll
    for (int r = 0; r < 8; ++r) {
      int q = qBase + r + 8 * hi;
      if (jb + lo > q)      s0[r] = -__builtin_inff();
      if (jb + 16 + lo > q) s1[r] = -__builtin_inff();
    }
    float rmax[8];
    #pragma unroll
    for (int r = 0; r < 8; ++r) rmax[r] = fmaxf(s0[r], s1[r]);
    #pragma unroll
    for (int msk = 1; msk < 16; msk <<= 1)
      #pragma unroll
      for (int r = 0; r < 8; ++r)
        rmax[r] = fmaxf(rmax[r], __shfl_xor(rmax[r], msk, 32));
    float alpha[8], rsum[8];
    #pragma unroll
    for (int r = 0; r < 8; ++r) {
      float mn = fmaxf(m[r], rmax[r]);
      alpha[r] = __expf(m[r] - mn);
      m[r] = mn;
      float p0 = __expf(s0[r] - mn);
      float p1 = __expf(s1[r] - mn);
      s0[r] = p0; s1[r] = p1;
      rsum[r] = p0 + p1;
    }
    #pragma unroll
    for (int msk = 1; msk < 16; msk <<= 1)
      #pragma unroll
      for (int r = 0; r < 8; ++r)
        rsum[r] += __shfl_xor(rsum[r], msk, 32);
    #pragma unroll
    for (int r = 0; r < 8; ++r) l[r] = l[r] * alpha[r] + rsum[r];
    #pragma unroll
    for (int dt = 0; dt < 8; ++dt)
      #pragma unroll
      for (int r = 0; r < 8; ++r) o[dt][r] *= alpha[r];

    // C-layout P -> LDS -> A-fragment layout
    #pragma unroll
    for (int r = 0; r < 8; ++r) {
      P[(r + 8 * hi) * PPITCH + lo]      = (__bf16)s0[r];
      P[(r + 8 * hi) * PPITCH + 16 + lo] = (__bf16)s1[r];
    }
    bf16x16 ap = frag16_split(&P[lo * PPITCH + hi * 8]);
    #pragma unroll
    for (int dt = 0; dt < 8; ++dt) {
      bf16x16 bv = frag16_contig(vplane + (long)(dt * 16 + lo) * S_ + jb + hi * 8);
      o[dt] = __builtin_amdgcn_wmma_f32_16x16x32_bf16(
          false, ap, false, bv, (short)0, o[dt], false, false);
    }
  }

  #pragma unroll
  for (int r = 0; r < 8; ++r) {
    float inv = (l[r] > 0.0f) ? 1.0f / l[r] : 0.0f;
    for (int dt = 0; dt < 8; ++dt) o[dt][r] *= inv;
  }
  #pragma unroll
  for (int dt = 0; dt < 8; ++dt)
    #pragma unroll
    for (int r = 0; r < 8; ++r) {
      int q = qBase + r + 8 * hi;
      attnb[((long)(b * S_ + q)) * DIM_ + h * HD_ + dt * 16 + lo] = (__bf16)o[dt][r];
    }
}

// =====================================================================
// Launch
// =====================================================================
extern "C" void kernel_launch(void* const* d_in, const int* in_sizes, int n_in,
                              void* d_out, int out_size, void* d_ws, size_t ws_size,
                              hipStream_t stream) {
  const float* x     = (const float*)d_in[0];
  const float* freqs = (const float*)d_in[2];
  const float* wq    = (const float*)d_in[5];
  const float* wk    = (const float*)d_in[6];
  const float* wv    = (const float*)d_in[7];
  const float* wo    = (const float*)d_in[8];

  float* out     = (float*)d_out;
  float* cache_k = out + OUT_ELEMS;
  float* cache_v = cache_k + CACHE_ELEMS;

  // workspace carve (~302 MB)
  float*  xq    = (float*)d_ws;                 // fp32 q before rope
  float*  xk    = xq + OUT_ELEMS;               // fp32 k before rope
  __bf16* xb    = (__bf16*)(xk + OUT_ELEMS);    // x in bf16
  __bf16* attnb = xb + OUT_ELEMS;               // attention out (bf16)
  __bf16* wbuf  = attnb + OUT_ELEMS;            // current weight^T (bf16)
  __bf16* kbuf  = wbuf + OUT_ELEMS;             // roped K (bf16)
  __bf16* vt    = kbuf + OUT_ELEMS;             // V^T (bf16) [b][h][d][s]
  __bf16* qbuf  = xb;                           // alias: xb dead after QKV GEMMs

  const dim3 gGrid(DIM_ / TN, (B_ * S_) / TM);   // 32 x 32
  const dim3 tGrid(DIM_ / 64, DIM_ / 64);        // 64 x 64
  const dim3 vGrid(S_ / 64, HD_ / 64, B_ * NH_); // 32 x 2 x 64
  const unsigned packBlocks = (unsigned)(OUT_ELEMS / (256 * 8));
  const unsigned zeroBlocks = (unsigned)((2L * S_ * DIM_) / (256 * 4));
  const unsigned ropeBlocks = (unsigned)(((long)B_ * S_ * NH_ * 64 + 255) / 256);

  pack_bf16_kernel<<<packBlocks, 256, 0, stream>>>(x, xb);

  transpose_pack_bf16_kernel<<<tGrid, 256, 0, stream>>>(wq, wbuf);
  gemm_bf16wmma<<<gGrid, 256, 0, stream>>>(xb, wbuf, xq);

  transpose_pack_bf16_kernel<<<tGrid, 256, 0, stream>>>(wk, wbuf);
  gemm_bf16wmma<<<gGrid, 256, 0, stream>>>(xb, wbuf, xk);

  transpose_pack_bf16_kernel<<<tGrid, 256, 0, stream>>>(wv, wbuf);
  gemm_bf16wmma<<<gGrid, 256, 0, stream>>>(xb, wbuf, cache_v);

  zero_tail_kernel<<<zeroBlocks, 256, 0, stream>>>(cache_k, cache_v);
  rope_cache_kernel<<<ropeBlocks, 256, 0, stream>>>(xq, xk, freqs, cache_k,
                                                    qbuf, kbuf);
  v_transpose_pack_kernel<<<vGrid, 256, 0, stream>>>(cache_v, vt);

  flash_attn_kernel<<<2048, 128, 0, stream>>>(qbuf, kbuf, vt, attnb);

  transpose_pack_bf16_kernel<<<tGrid, 256, 0, stream>>>(wo, wbuf);
  gemm_bf16wmma<<<gGrid, 256, 0, stream>>>(attnb, wbuf, out);
}
